// LearnableHarmonicSynth_28174985461987
// MI455X (gfx1250) — matrix operand order repaired and checked by reading, not compiled, not run on hardware
//
#include <hip/hip_runtime.h>

typedef __attribute__((ext_vector_type(16))) _Float16 v16h;
typedef __attribute__((ext_vector_type(8)))  float    v8f;

#define SR_F        44100.0f
#define NYQ_F       22050.0f
#define TWO_PI_APX  6.28318f
#define NYQ_SCALE   1.0e-4f
#define INV_2PI_D   0.15915494309189535   // 1/(2*pi), double

// coefficient for harmonic multiplier h (1-based): h==1 -> 1.0 (fundamental),
// h>=2 -> scale(h) * harmonic_amplitudes[h-1]
__device__ __forceinline__ float harm_coeff(int h, float freq_hz, const float* __restrict__ amp) {
    if (h == 1) return 1.0f;
    float scale = (freq_hz * (float)h > NYQ_F) ? NYQ_SCALE : 1.0f;
    return scale * amp[h - 1];
}

// sin/cos(2*pi*(K*id + phi)) via f64 phase reduction + hardware trans ops
// (v_sin_f32 / v_cos_f32 take input pre-scaled by 1/2pi, i.e. revolutions).
__device__ __forceinline__ float sin_rev(double K, double phi, double id) {
    double r = __builtin_fma(K, id, phi);
    r -= __builtin_rint(r);
    return __builtin_amdgcn_sinf((float)r);
}
__device__ __forceinline__ float cos_rev(double K, double phi, double id) {
    double r = __builtin_fma(K, id, phi);
    r -= __builtin_rint(r);
    return __builtin_amdgcn_cosf((float)r);
}

__global__ void __launch_bounds__(256)
harmonic_synth_wmma(const float* __restrict__ freq_p,
                    const float* __restrict__ phase_p,
                    const float* __restrict__ amp,
                    float* __restrict__ out,
                    int L) {
    const int  lane     = threadIdx.x & 31;
    const int  waveId   = (blockIdx.x * (blockDim.x >> 5)) + (threadIdx.x >> 5);
    const int  numWaves = gridDim.x * (blockDim.x >> 5);
    const int  numTiles = (L + 255) >> 8;          // 256 samples per tile
    const int  n        = lane & 15;
    const bool hi       = lane >= 16;

    const float  freq    = freq_p[0];
    const float  phase   = phase_p[0];
    const float  freq_hz = freq * SR_F / TWO_PI_APX;
    // x_i = freq * L/(L-1) * i  (linspace endpoint-inclusive)
    const double step_d  = (double)freq * ((double)L / (double)(L - 1));
    const double revs    = step_d * INV_2PI_D;     // revolutions/sample for h=1
    const double phiR    = (double)phase * INV_2PI_D;

    // ---- tile-invariant B matrices: B1[k][n]=cos(h*step*n), B2[k][n]=sin(h*step*n) ----
    // B 32x16 f16 layout: lanes 0-15 hold K={0..7,16..23} of column n=lane; lanes 16-31 K={8..15,24..31}.
    // Per lane+chunk, two runs of 8 contiguous h: h = hb+32c+16q+t  -> Chebyshev recurrence over h.
    v16h b1c0, b2c0, b1c1, b2c1;
    {
        const double nd  = (double)n;
        const int    hb  = hi ? 9 : 1;
        const float  c2n = 2.0f * cos_rev(revs, 0.0, nd);    // 2*cos(step*n)
#pragma unroll
        for (int c = 0; c < 2; ++c) {
            v16h& b1 = c ? b1c1 : b1c0;
            v16h& b2 = c ? b2c1 : b2c0;
#pragma unroll
            for (int q = 0; q < 2; ++q) {
            const double Kh = revs * (double)(hb + 32 * c + 16 * q);
            float Sp = sin_rev(Kh,        0.0, nd);
            float Sc = sin_rev(Kh + revs, 0.0, nd);
            float Cp = cos_rev(Kh,        0.0, nd);
            float Cc = cos_rev(Kh + revs, 0.0, nd);
            b1[8 * q] = (_Float16)Cp;  b1[8 * q + 1] = (_Float16)Cc;
            b2[8 * q] = (_Float16)Sp;  b2[8 * q + 1] = (_Float16)Sc;
#pragma unroll
            for (int j = 2; j < 8; ++j) {
                float Sn = fmaf(c2n, Sc, -Sp);  Sp = Sc;  Sc = Sn;
                float Cn = fmaf(c2n, Cc, -Cp);  Cp = Cc;  Cc = Cn;
                b1[8 * q + j] = (_Float16)Cn;
                b2[8 * q + j] = (_Float16)Sn;
            }
            }
        }
    }

    // ---- coefficient packs in A layout (lane<16: K={0..7,16..23}; lane>=16: K={8..15,24..31}) ----
    v16h cf0, cf1;
    {
        const int kbase = hi ? 8 : 0;
#pragma unroll
        for (int j = 0; j < 16; ++j) {
            const int K = kbase + ((j < 8) ? j : j + 8);
            cf0[j] = (_Float16)harm_coeff(K + 1,  freq_hz, amp);   // chunk 0: h = 1..32
            cf1[j] = (_Float16)harm_coeff(K + 33, freq_hz, amp);   // chunk 1: h = 33..64
        }
    }

    const int hbase = hi ? 9 : 1;   // first harmonic of this lane's run r=0 (A layout)

    for (int tile = waveId; tile < numTiles; tile += numWaves) {   // wave-uniform loop
        const int    base = tile << 8;
        const int    srow = base + (n << 4);        // sample index of A row M = lane&15
        const double sd   = (double)srow;
        const float  c2X  = 2.0f * cos_rev(revs, 0.0, sd);   // 2*cos(X), X = step*srow

        // run-0 seeds (only f64-reduced trig this tile) + the run-jump rotation by D=16*X
        const double K0 = revs * (double)hbase;
        float Sp = sin_rev(K0,        phiR, sd);
        float Sc = sin_rev(K0 + revs, phiR, sd);
        float Cp = cos_rev(K0,        phiR, sd);
        float Cc = cos_rev(K0 + revs, phiR, sd);
        const float cD = cos_rev(revs * 16.0, 0.0, sd);      // cos(16X)
        const float sD = sin_rev(revs * 16.0, 0.0, sd);      // sin(16X)

        // A trig families: 4 runs of 8 contiguous h (Chebyshev within run, rotate between runs)
        v16h s0h, c0h, s1h, c1h;
#pragma unroll
        for (int r = 0; r < 4; ++r) {
            const int jo = (r & 1) ? 8 : 0;
            v16h& sh = (r < 2) ? s0h : s1h;
            v16h& ch = (r < 2) ? c0h : c1h;
            sh[jo]     = (_Float16)Sp;  sh[jo + 1] = (_Float16)Sc;
            ch[jo]     = (_Float16)Cp;  ch[jo + 1] = (_Float16)Cc;
            float Sa = Sp, Sb = Sc, Ca = Cp, Cb = Cc;
#pragma unroll
            for (int t = 2; t < 8; ++t) {
                float Sn = fmaf(c2X, Sb, -Sa);  Sa = Sb;  Sb = Sn;
                float Cn = fmaf(c2X, Cb, -Ca);  Ca = Cb;  Cb = Cn;
                sh[jo + t] = (_Float16)Sn;
                ch[jo + t] = (_Float16)Cn;
            }
            if (r < 3) {   // rotate seeds by D=16X: sin(a+D)=sa*cD+ca*sD, cos(a+D)=ca*cD-sa*sD
                float Sp2 = fmaf(Sp, cD,  Cp * sD);
                float Cp2 = fmaf(Cp, cD, -(Sp * sD));
                float Sc2 = fmaf(Sc, cD,  Cc * sD);
                float Cc2 = fmaf(Cc, cD, -(Sc * sD));
                Sp = Sp2;  Cp = Cp2;  Sc = Sc2;  Cc = Cc2;
            }
        }

        // fold coefficients in with packed f16 multiplies (v_pk_mul_f16)
        const v16h a1c0 = cf0 * s0h;   // c_h * sin(a)
        const v16h a2c0 = cf0 * c0h;   // c_h * cos(a)
        const v16h a1c1 = cf1 * s1h;
        const v16h a2c1 = cf1 * c1h;

        // D = A1*B1 + A2*B2 over both K-chunks: 4 chained WMMAs -> all 256 samples
        v8f acc = {};
        acc = __builtin_amdgcn_wmma_f32_16x16x32_f16(false, a1c0, false, b1c0, (short)0, acc, false, false);
        acc = __builtin_amdgcn_wmma_f32_16x16x32_f16(false, a2c0, false, b2c0, (short)0, acc, false, false);
        acc = __builtin_amdgcn_wmma_f32_16x16x32_f16(false, a1c1, false, b1c1, (short)0, acc, false, false);
        acc = __builtin_amdgcn_wmma_f32_16x16x32_f16(false, a2c1, false, b2c1, (short)0, acc, false, false);

        // D element e: M = e + (hi?8:0), N = lane&15  ->  sample = base + 16*M + N
#pragma unroll
        for (int e = 0; e < 8; ++e) {
            const int s = base + ((e + (hi ? 8 : 0)) << 4) + n;
            if (s < L) out[s] = acc[e];
        }
    }
}

extern "C" void kernel_launch(void* const* d_in, const int* in_sizes, int n_in,
                              void* d_out, int out_size, void* d_ws, size_t ws_size,
                              hipStream_t stream) {
    // inputs (setup_inputs order): freq f32[1], output_length_samples int[1],
    // gain f32[1] (unused in fwd), phase f32[1], harmonic_amplitudes f32[64]
    const float* freq  = (const float*)d_in[0];
    const float* phase = (const float*)d_in[3];
    const float* amp   = (const float*)d_in[4];
    float* out = (float*)d_out;

    const int L = out_size;                 // == output_length_samples
    const int tiles = (L + 255) / 256;      // one 256-sample tile per wave
    const int wavesPerBlock = 8;            // 256 threads, wave32
    const int blocks = (tiles + wavesPerBlock - 1) / wavesPerBlock;

    harmonic_synth_wmma<<<blocks, 256, 0, stream>>>(freq, phase, amp, out, L);
}
